// GenesisAttention_61100204753403
// MI455X (gfx1250) — matrix operand order repaired and checked by reading, not compiled, not run on hardware
//
#include <hip/hip_runtime.h>
#include <hip/hip_bf16.h>

// ---------------------------------------------------------------------------
// Problem constants (must match reference)
// ---------------------------------------------------------------------------
#define BATCH      2
#define SEQ        2048
#define DMODEL     2048
#define NHQ        16
#define NHKV       4
#define HDIM       128
#define KVW        (NHKV * HDIM)      // 512
#define WINDOW     512
#define ROWS       (BATCH * SEQ)      // 4096
#define SOFT_SCALE 0.08838834764831845f   // 1/sqrt(128)
#define NEGBIG     (-1.0e30f)

typedef __attribute__((ext_vector_type(16))) __bf16 v16bf;
typedef __attribute__((ext_vector_type(8)))  __bf16 v8bf;
typedef __attribute__((ext_vector_type(8)))  float  v8f;

// ---------------------------------------------------------------------------
// gfx1250 async global->LDS copy (ASYNCcnt-tracked), with sync fallback.
// Builtin prototype (from hipcc diagnostic): (int4 AS1*, int4 AS3*, Ii, Ii).
// ---------------------------------------------------------------------------
#if __has_builtin(__builtin_amdgcn_global_load_async_to_lds_b128)
#define HAVE_ASYNC_LDS 1
#else
#define HAVE_ASYNC_LDS 0
#endif

typedef int vi4 __attribute__((vector_size(16)));
typedef __attribute__((address_space(1))) vi4 as1_vi4;
typedef __attribute__((address_space(3))) vi4 as3_vi4;

static __device__ __forceinline__ void cp16_g2l(const __bf16* g, __bf16* l) {
#if HAVE_ASYNC_LDS
    vi4* gp = (vi4*)(void*)const_cast<__bf16*>(g);
    vi4* lp = (vi4*)(void*)l;
    __builtin_amdgcn_global_load_async_to_lds_b128((as1_vi4*)gp, (as3_vi4*)lp, 0, 0);
#else
    *(v8bf*)l = *(const v8bf*)g;
#endif
}

#if HAVE_ASYNC_LDS
#if __has_builtin(__builtin_amdgcn_s_wait_asynccnt)
#define WAIT_ASYNC(n) __builtin_amdgcn_s_wait_asynccnt(n)
#else
#define WAIT_ASYNC(n) asm volatile("s_wait_asynccnt %0" ::"i"(n))
#endif
#else
#define WAIT_ASYNC(n)
#endif

static __device__ __forceinline__ v8f zero8() {
    v8f z;
#pragma unroll
    for (int i = 0; i < 8; ++i) z[i] = 0.0f;
    return z;
}

static __device__ __forceinline__ v16bf mk16(v8bf lo, v8bf hi) {
    v16bf r;
#pragma unroll
    for (int i = 0; i < 8; ++i) { r[i] = lo[i]; r[i + 8] = hi[i]; }
    return r;
}

// D = A(16x32 bf16) * B(32x16 bf16) + C(16x16 f32)
static __device__ __forceinline__ v8f wmma_bf(v16bf a, v16bf b, v8f c) {
    return __builtin_amdgcn_wmma_f32_16x16x32_bf16(
        /*neg_a=*/false, a, /*neg_b=*/false, b,
        /*c_mod=*/(short)0, c, /*reuse_a=*/false, /*reuse_b=*/false);
}

// ---------------------------------------------------------------------------
// fp32 -> bf16 streaming convert (activations)
// ---------------------------------------------------------------------------
__global__ __launch_bounds__(256) void cvt_f32_bf16(const float* __restrict__ s,
                                                    __bf16* __restrict__ d,
                                                    long n) {
    long i = (long)blockIdx.x * blockDim.x + threadIdx.x;
    long stride = (long)gridDim.x * blockDim.x;
    for (; i < n; i += stride) d[i] = (__bf16)s[i];
}

// ---------------------------------------------------------------------------
// fp32 -> bf16 convert + transpose for weights: W[K][N] -> WT[N][K].
// WT rows are contiguous in K => GEMM B-tiles become plain row copies
// (async-LDS friendly, no scatter). Writes are coalesced.
// ---------------------------------------------------------------------------
__global__ __launch_bounds__(256) void cvtT_f32_bf16(const float* __restrict__ W,
                                                     __bf16* __restrict__ WT,
                                                     int K, int N) {
    long total = (long)K * N;
    long i = (long)blockIdx.x * blockDim.x + threadIdx.x;
    long stride = (long)gridDim.x * blockDim.x;
    for (; i < total; i += stride) {
        int n = (int)(i / K);
        int k = (int)(i % K);
        WT[i] = (__bf16)W[(size_t)k * N + n];
    }
}

// ---------------------------------------------------------------------------
// Tiled bf16 WMMA GEMM: C[M,N] = A[M,K] * Bt[N,K]^T
// 256 threads (8 waves, 4x2); block tile 128x128, BK=32, double-buffered LDS
// filled with per-lane async b128 copies. 8 v_wmma per wave per k-step.
// ---------------------------------------------------------------------------
#define GBM 128
#define GBN 128
#define GBK 32

__global__ __launch_bounds__(256) void gemm_bf16_wmma(
    const __bf16* __restrict__ A, const __bf16* __restrict__ Bt,
    __bf16* __restrict__ Cb, float* __restrict__ Cf,
    int M, int N, int K) {
    __shared__ __bf16 sA[2][GBM * GBK];   // [row][k]
    __shared__ __bf16 sB[2][GBN * GBK];   // [n][k]

    const int tid  = threadIdx.x;
    const int lane = tid & 31;
    const int w    = tid >> 5;
    const int wm   = w & 3;               // 4 wave-rows of 32
    const int wn   = w >> 2;              // 2 wave-cols of 64
    const int mBase = blockIdx.y * GBM;
    const int nBase = blockIdx.x * GBN;

    // Each tile = 512 chunks of 16B per operand; each thread owns 2+2 chunks
    // => 4 async instructions per wave per tile.
    auto issue = [&](int k0, int buf) {
#pragma unroll
        for (int j = 0; j < 2; ++j) {
            int c   = tid * 2 + j;        // 0..511
            int row = c >> 2;             // 4 chunks per 32-elem row
            int kc  = (c & 3) * 8;
            cp16_g2l(&A[(size_t)(mBase + row) * K + k0 + kc],
                     &sA[buf][row * GBK + kc]);
        }
#pragma unroll
        for (int j = 0; j < 2; ++j) {
            int c   = tid * 2 + j;
            int nr  = c >> 2;
            int kc  = (c & 3) * 8;
            cp16_g2l(&Bt[(size_t)(nBase + nr) * K + k0 + kc],
                     &sB[buf][nr * GBK + kc]);
        }
    };

    v8f acc[2][4];
#pragma unroll
    for (int mi = 0; mi < 2; ++mi)
#pragma unroll
        for (int ni = 0; ni < 4; ++ni) acc[mi][ni] = zero8();

    issue(0, 0);
    const int nk = K / GBK;
    for (int kt = 0; kt < nk; ++kt) {
        const int cur = kt & 1;
        if (kt + 1 < nk) {
            issue((kt + 1) * GBK, cur ^ 1);   // prefetch next tile (async)
            WAIT_ASYNC(4);                    // current tile landed; next in flight
        } else {
            WAIT_ASYNC(0);
        }
        __syncthreads();

        const __bf16* a  = sA[cur];
        const __bf16* bt = sB[cur];
        const int kb8  = (lane >> 4) * 8;
        const int kb16 = (lane >> 4) * 16;    // B striping
        v16bf af[2], bfv[4];
#pragma unroll
        for (int mi = 0; mi < 2; ++mi) {
            int row = wm * 32 + mi * 16 + (lane & 15);
            af[mi] = mk16(*(const v8bf*)&a[row * GBK + kb16],
                          *(const v8bf*)&a[row * GBK + kb16 + 8]);
        }
#pragma unroll
        for (int ni = 0; ni < 4; ++ni) {
            int col = wn * 64 + ni * 16 + (lane & 15);
            bfv[ni] = *(const v16bf*)&bt[col * GBK + kb16];
        }
#pragma unroll
        for (int mi = 0; mi < 2; ++mi)
#pragma unroll
            for (int ni = 0; ni < 4; ++ni)
                acc[mi][ni] = wmma_bf(af[mi], bfv[ni], acc[mi][ni]);
        __syncthreads();                      // protect cur buf before reuse
    }

    // ---- epilogue: lane holds col (lane&15), rows r + 8*(lane>>4)
#pragma unroll
    for (int mi = 0; mi < 2; ++mi) {
#pragma unroll
        for (int ni = 0; ni < 4; ++ni) {
            int row0 = mBase + wm * 32 + mi * 16 + 8 * (lane >> 4);
            int col  = nBase + wn * 64 + ni * 16 + (lane & 15);
#pragma unroll
            for (int r = 0; r < 8; ++r) {
                float v = acc[mi][ni][r];
                size_t idx = (size_t)(row0 + r) * N + col;
                if (Cf) Cf[idx] = v;
                else    Cb[idx] = (__bf16)v;
            }
        }
    }
}

// ---------------------------------------------------------------------------
// RoPE (interleaved pairs), in-place on bf16 activations, fp32 math
// ---------------------------------------------------------------------------
__global__ __launch_bounds__(256) void rope_kernel(__bf16* __restrict__ X,
                                                   const float* __restrict__ cs,
                                                   const float* __restrict__ sn,
                                                   int H, long total) {
    long idx = (long)blockIdx.x * blockDim.x + threadIdx.x;
    long stride = (long)gridDim.x * blockDim.x;
    for (; idx < total; idx += stride) {
        int  i  = (int)(idx & 63);
        int  h  = (int)((idx >> 6) % H);
        long bt = idx / (64L * H);
        int  t  = (int)(bt % SEQ);
        __bf16* p = X + (size_t)bt * (H * HDIM) + h * HDIM + 2 * i;
        float c = cs[t * 64 + i], s = sn[t * 64 + i];
        float e = (float)p[0], o = (float)p[1];
        p[0] = (__bf16)(e * c - o * s);
        p[1] = (__bf16)(e * s + o * c);
    }
}

// ---------------------------------------------------------------------------
// V transpose: (b,t,kvh,d) -> (b,kvh,d,t) so PV B-frags are contiguous loads
// ---------------------------------------------------------------------------
__global__ __launch_bounds__(256) void vtrans_kernel(const __bf16* __restrict__ V,
                                                     __bf16* __restrict__ VT) {
    long idx = (long)blockIdx.x * blockDim.x + threadIdx.x;
    long stride = (long)gridDim.x * blockDim.x;
    const long total = (long)ROWS * KVW;
    for (; idx < total; idx += stride) {
        int  d   = (int)(idx & 127);
        int  kvh = (int)((idx >> 7) & 3);
        long bt  = idx >> 9;
        int  t   = (int)(bt % SEQ);
        int  b   = (int)(bt / SEQ);
        VT[((size_t)(b * NHKV + kvh) * HDIM + d) * SEQ + t] = V[idx];
    }
}

// ---------------------------------------------------------------------------
// Flash attention, sliding window 512, GQA 4:1.
// One wave per (b, h, 16-query tile). Key tiles of 32 -> per tile:
// 8 WMMA for S = Q*K^T (contraction over head dim 128) and
// 8 WMMA for O += P*V (contraction over 32 keys).
// ---------------------------------------------------------------------------
__global__ __launch_bounds__(32) void attn_kernel(const __bf16* __restrict__ Q,
                                                  const __bf16* __restrict__ Km,
                                                  const __bf16* __restrict__ VT,
                                                  __bf16* __restrict__ O) {
    __shared__ __bf16 sP[16 * 32];      // P tile staging (C-layout -> A-layout)

    const int lane = threadIdx.x & 31;
    const int qb   = blockIdx.x * 16;
    const int h    = blockIdx.y;
    const int b    = blockIdx.z;
    const int kvh  = h >> 2;
    const int kcol = lane & 15;
    const int kb8  = (lane >> 4) * 8;
    const int kb16 = (lane >> 4) * 16;

    const __bf16* Qrow = Q + ((size_t)(b * SEQ) + qb + kcol) * DMODEL + h * HDIM;
    v16bf qf[4];
#pragma unroll
    for (int f = 0; f < 4; ++f)
        qf[f] = mk16(*(const v8bf*)&Qrow[f * 32 + kb16],
                     *(const v8bf*)&Qrow[f * 32 + kb16 + 8]);

    const __bf16* Kbase  = Km + (size_t)b * SEQ * KVW + kvh * HDIM;
    const __bf16* VTbase = VT + (size_t)(b * NHKV + kvh) * HDIM * SEQ;

    float m[8], l[8];
    v8f o[8];
#pragma unroll
    for (int r = 0; r < 8; ++r) { m[r] = NEGBIG; l[r] = 0.0f; }
#pragma unroll
    for (int nf = 0; nf < 8; ++nf) o[nf] = zero8();

    int ks = qb - (WINDOW - 1);
    if (ks < 0) ks = 0;
    ks &= ~31;
    const int ke = (qb + 15) & ~31;

    for (int kt = ks; kt <= ke; kt += 32) {
        // ---- S = Q K^T (K rows contiguous in d -> direct B-frags)
        v8f s0 = zero8(), s1 = zero8();
#pragma unroll
        for (int f = 0; f < 4; ++f) {
            v16bf k0 = *(const v16bf*)&Kbase[(size_t)(kt + kcol)      * KVW + f * 32 + kb16];
            v16bf k1 = *(const v16bf*)&Kbase[(size_t)(kt + 16 + kcol) * KVW + f * 32 + kb16];
            s0 = wmma_bf(qf[f], k0, s0);
            s1 = wmma_bf(qf[f], k1, s1);
        }

        // ---- mask + online softmax (16-lane butterfly row reductions)
        float alpha[8];
#pragma unroll
        for (int r = 0; r < 8; ++r) {
            int q    = qb + r + 8 * (lane >> 4);
            int key0 = kt + kcol;
            int key1 = key0 + 16;
            float a0 = s0[r] * SOFT_SCALE;
            float a1 = s1[r] * SOFT_SCALE;
            a0 = ((key0 <= q) && (q - key0 < WINDOW)) ? a0 : NEGBIG;
            a1 = ((key1 <= q) && (q - key1 < WINDOW)) ? a1 : NEGBIG;
            float mx = fmaxf(a0, a1);
#pragma unroll
            for (int x = 1; x < 16; x <<= 1) mx = fmaxf(mx, __shfl_xor(mx, x));
            float mn = fmaxf(m[r], mx);
            alpha[r] = __expf(m[r] - mn);
            m[r] = mn;
            float p0 = __expf(a0 - mn);
            float p1 = __expf(a1 - mn);
            s0[r] = p0; s1[r] = p1;
            float rs = p0 + p1;
#pragma unroll
            for (int x = 1; x < 16; x <<= 1) rs += __shfl_xor(rs, x);
            l[r] = l[r] * alpha[r] + rs;
        }
#pragma unroll
        for (int nf = 0; nf < 8; ++nf)
#pragma unroll
            for (int r = 0; r < 8; ++r) o[nf][r] *= alpha[r];

        // ---- re-stripe P from C-layout into A-layout via LDS
#pragma unroll
        for (int r = 0; r < 8; ++r) {
            int row = r + 8 * (lane >> 4);
            sP[row * 32 + kcol]      = (__bf16)s0[r];
            sP[row * 32 + 16 + kcol] = (__bf16)s1[r];
        }
        __syncthreads();
        v16bf pf = mk16(*(const v8bf*)&sP[kcol * 32 + kb16],
                        *(const v8bf*)&sP[kcol * 32 + kb16 + 8]);
        __syncthreads();

        // ---- O += P * V (VT rows contiguous in t -> direct B-frags)
#pragma unroll
        for (int nf = 0; nf < 8; ++nf) {
            v16bf vf = *(const v16bf*)&VTbase[(size_t)(nf * 16 + kcol) * SEQ + kt + kb16];
            o[nf] = wmma_bf(pf, vf, o[nf]);
        }
    }

    // ---- finalize: divide by row sums, store bf16 (b*T+q, h*128+d)
    float inv[8];
#pragma unroll
    for (int r = 0; r < 8; ++r) inv[r] = 1.0f / l[r];
#pragma unroll
    for (int nf = 0; nf < 8; ++nf) {
#pragma unroll
        for (int r = 0; r < 8; ++r) {
            int q   = qb + r + 8 * (lane >> 4);
            int col = h * HDIM + nf * 16 + kcol;
            O[(size_t)(b * SEQ + q) * DMODEL + col] = (__bf16)(o[nf][r] * inv[r]);
        }
    }
}

// ---------------------------------------------------------------------------
// Host-side orchestration
// ---------------------------------------------------------------------------
extern "C" void kernel_launch(void* const* d_in, const int* in_sizes, int n_in,
                              void* d_out, int out_size, void* d_ws, size_t ws_size,
                              hipStream_t stream) {
    (void)in_sizes; (void)n_in; (void)out_size; (void)ws_size;
    const float* x    = (const float*)d_in[0];
    const float* fcos = (const float*)d_in[1];
    const float* fsin = (const float*)d_in[2];
    // d_in[3] = bool mask: unused (window computed analytically)
    const float* wq   = (const float*)d_in[4];
    const float* wk   = (const float*)d_in[5];
    const float* wv   = (const float*)d_in[6];
    const float* wo   = (const float*)d_in[7];

    char* ws = (char*)d_ws;
    const size_t MB = 1024 * 1024;
    __bf16* xb  = (__bf16*)(ws + 0);        // 16 MB  (4096x2048)
    __bf16* wqT = (__bf16*)(ws + 16 * MB);  //  8 MB  (2048x2048) [n][k]
    __bf16* wkT = (__bf16*)(ws + 24 * MB);  //  2 MB  (512x2048)  [n][k]
    __bf16* wvT = (__bf16*)(ws + 26 * MB);  //  2 MB  (512x2048)  [n][k]
    __bf16* woT = (__bf16*)(ws + 28 * MB);  //  8 MB  (2048x2048) [n][k]
    __bf16* Qb  = (__bf16*)(ws + 36 * MB);  // 16 MB  (4096x2048)
    __bf16* Kb  = (__bf16*)(ws + 52 * MB);  //  4 MB  (4096x512)
    __bf16* Vb  = (__bf16*)(ws + 56 * MB);  //  4 MB  (4096x512)
    // Reuse: VT overwrites wqT (only needed for Q projection, done by then);
    //        attention output overwrites xb (only needed for projections).
    __bf16* VTb = wqT;                      //  4 MB  (2,4,128,2048)
    __bf16* Ob  = xb;                       // 16 MB  (4096x2048)

    // 1) fp32 -> bf16 conversions (weights transposed to [n][k])
    cvt_f32_bf16 <<<4096, 256, 0, stream>>>(x,  xb,  (long)ROWS * DMODEL);
    cvtT_f32_bf16<<<2048, 256, 0, stream>>>(wq, wqT, DMODEL, DMODEL);
    cvtT_f32_bf16<<<1024, 256, 0, stream>>>(wk, wkT, DMODEL, KVW);
    cvtT_f32_bf16<<<1024, 256, 0, stream>>>(wv, wvT, DMODEL, KVW);
    cvtT_f32_bf16<<<2048, 256, 0, stream>>>(wo, woT, DMODEL, DMODEL);

    // 2) QKV projections
    gemm_bf16_wmma<<<dim3(DMODEL / GBN, ROWS / GBM), 256, 0, stream>>>(
        xb, wqT, Qb, nullptr, ROWS, DMODEL, DMODEL);
    gemm_bf16_wmma<<<dim3(KVW / GBN, ROWS / GBM), 256, 0, stream>>>(
        xb, wkT, Kb, nullptr, ROWS, KVW, DMODEL);
    gemm_bf16_wmma<<<dim3(KVW / GBN, ROWS / GBM), 256, 0, stream>>>(
        xb, wvT, Vb, nullptr, ROWS, KVW, DMODEL);

    // 3) RoPE on Q and K
    rope_kernel<<<4096, 256, 0, stream>>>(Qb, fcos, fsin, NHQ,  (long)ROWS * NHQ  * 64);
    rope_kernel<<<2048, 256, 0, stream>>>(Kb, fcos, fsin, NHKV, (long)ROWS * NHKV * 64);

    // 4) V transpose for contiguous PV fragments
    vtrans_kernel<<<2048, 256, 0, stream>>>(Vb, VTb);

    // 5) Sliding-window flash attention
    attn_kernel<<<dim3(SEQ / 16, NHQ, BATCH), 32, 0, stream>>>(Qb, Kb, VTb, Ob);

    // 6) Output projection -> fp32 d_out
    gemm_bf16_wmma<<<dim3(DMODEL / GBN, ROWS / GBM), 256, 0, stream>>>(
        Ob, woT, nullptr, (float*)d_out, ROWS, DMODEL, DMODEL);
}
